// GptOssMLP_55173149884959
// MI455X (gfx1250) — matrix operand order repaired and checked by reading, not compiled, not run on hardware
//
#include <hip/hip_runtime.h>
#include <stdint.h>

// ---------------- problem constants ----------------
constexpr int   Tt    = 512;     // tokens (B*S)
constexpr int   Hh    = 2880;    // hidden
constexpr int   Ii    = 2880;    // intermediate
constexpr int   Ee    = 16;      // experts
constexpr int   TOPK  = 4;
constexpr float ALPHA = 1.702f;
constexpr float LIMIT = 7.0f;

// ---------------- tiling ----------------
constexpr int BM  = 64;   // tokens per block tile (4 wmma M-subtiles)
constexpr int BK  = 32;   // K per wmma step
constexpr int BND = 64;   // gate_up d-columns per block (= 32 i-columns)
constexpr int BNI = 32;   // i-columns per block
constexpr int BNH = 64;   // down h-columns per block
constexpr int LDA = BK + 2;   // padded LDS row stride (bf16) -> 68B, conflict-friendly
constexpr int MTILES = Tt / BM;   // 8 worst-case m-tiles per expert

typedef __attribute__((ext_vector_type(16))) __bf16   v16bf;
typedef __attribute__((ext_vector_type(2)))  __bf16   bf16x2;
typedef __attribute__((ext_vector_type(8)))  float    v8f;
typedef __attribute__((ext_vector_type(2)))  float    f32x2;
typedef __attribute__((ext_vector_type(4)))  unsigned u32x4;
typedef __attribute__((ext_vector_type(8)))  int      i32x8;
typedef __attribute__((ext_vector_type(4)))  int      i32x4;

// ---------------- workspace layout (bytes) ----------------
constexpr size_t OFF_TOPI = 0;                                   // int  [T][4]
constexpr size_t OFF_TOPV = OFF_TOPI + (size_t)Tt * TOPK * 4;    // f32  [T][4]
constexpr size_t OFF_CNT  = OFF_TOPV + (size_t)Tt * TOPK * 4;    // int  [E]
constexpr size_t OFF_BASE = OFF_CNT + 64;                        // int  [E]
constexpr size_t OFF_TOK  = OFF_BASE + 64;                       // int  [2048] token id per slot
constexpr size_t OFF_SLOT = OFF_TOK + (size_t)Tt * TOPK * 4;     // int  [T][4] slot per (t,k)
constexpr size_t OFF_ACT  = ((OFF_SLOT + (size_t)Tt * TOPK * 4) + 255) & ~(size_t)255;
// act:  bf16 [2048][I]   = 11.8 MB
constexpr size_t OFF_EOUT = OFF_ACT + (size_t)Tt * TOPK * Ii * 2;
// eout: f32  [2048][H]   = 23.6 MB

// ======================================================================
// helpers
// ======================================================================
__device__ inline unsigned pack_bf16(float a, float b) {
  union { bf16x2 v; unsigned u; } p;
  p.v[0] = (__bf16)a; p.v[1] = (__bf16)b;   // -> v_cvt_pk_bf16_f32
  return p.u;
}

// WMMA fragment gathers (layouts per CDNA5 ISA 7.12.2, bf16 16x16x32)
// A (16x32): lane l holds row M=l%16; VGPR v pair j -> K=(v%4)*2+j + (l/16)*8 + (v/4)*16
// B (32x16): lane l holds col N=l%16; VGPR v pair j -> K=(v%4)*2+j + (l/16)*8 + (v/4)*16
__device__ inline v16bf frag_a_lds(const __bf16* rowbase, int khalf) {
  union { v16bf v; unsigned u[8]; } f;
#pragma unroll
  for (int vv = 0; vv < 8; ++vv) {
    int k = (vv & 3) * 2 + khalf * 8 + (vv >> 2) * 16;
    f.u[vv] = *(const unsigned*)(rowbase + k);   // dword-aligned: k even, stride 68B
  }
  return f.v;
}
__device__ inline v16bf frag_b_lds(const __bf16* colbase, int khalf) {
  union { v16bf v; unsigned u[8]; } f;
#pragma unroll
  for (int vv = 0; vv < 8; ++vv) {
    int k = (vv & 3) * 2 + khalf * 8 + (vv >> 2) * 16;
    f.u[vv] = *(const unsigned*)(colbase + k);
  }
  return f.v;
}

// ======================================================================
// Kernel 1: router — logits, top-4, softmax
// ======================================================================
__global__ __launch_bounds__(128) void router_kernel(
    const float* __restrict__ x,   // [T][H]
    const float* __restrict__ rw,  // [H][E]
    const float* __restrict__ rb,  // [E]
    int* __restrict__ topi,        // [T][4]
    float* __restrict__ topv)      // [T][4]
{
  const int t   = blockIdx.x;
  const int tid = threadIdx.x;
  const float* xr = x + (size_t)t * Hh;

  float acc[Ee];
#pragma unroll
  for (int e = 0; e < Ee; ++e) acc[e] = 0.f;

  for (int h = tid; h < Hh; h += 128) {
    float xv = xr[h];
    const float* w = rw + (size_t)h * Ee;
#pragma unroll
    for (int e = 0; e < Ee; ++e) acc[e] += xv * w[e];
  }

  __shared__ float red[Ee][128];
#pragma unroll
  for (int e = 0; e < Ee; ++e) red[e][tid] = acc[e];
  __syncthreads();

  __shared__ float logits[Ee];
  if (tid < Ee) {
    float s = rb[tid];
    for (int j = 0; j < 128; ++j) s += red[tid][j];
    logits[tid] = s;
  }
  __syncthreads();

  if (tid == 0) {
    float lg[Ee];
#pragma unroll
    for (int e = 0; e < Ee; ++e) lg[e] = logits[e];
    int   bi[TOPK]; float bv[TOPK];
    bool used[Ee];
#pragma unroll
    for (int e = 0; e < Ee; ++e) used[e] = false;
    for (int k = 0; k < TOPK; ++k) {          // ties -> first index (matches lax.top_k)
      float best = -3.4e38f; int idx = 0;
      for (int e = 0; e < Ee; ++e)
        if (!used[e] && lg[e] > best) { best = lg[e]; idx = e; }
      used[idx] = true; bi[k] = idx; bv[k] = best;
    }
    float mx = bv[0], s = 0.f;
#pragma unroll
    for (int k = 0; k < TOPK; ++k) { bv[k] = __expf(bv[k] - mx); s += bv[k]; }
    float inv = 1.f / s;
#pragma unroll
    for (int k = 0; k < TOPK; ++k) {
      topi[t * TOPK + k] = bi[k];
      topv[t * TOPK + k] = bv[k] * inv;
    }
  }
}

// ======================================================================
// Kernel 2: deterministic per-expert compaction (token order preserved)
// ======================================================================
__global__ void build_lists_kernel(
    const int* __restrict__ topi,
    int* __restrict__ counts, int* __restrict__ base,
    int* __restrict__ tok_list, int* __restrict__ slot_of)
{
  __shared__ int cnt_s[Ee];
  const int e = threadIdx.x;           // blockDim == 16
  if (e < Ee) {
    int c = 0;
    for (int t = 0; t < Tt; ++t)
#pragma unroll
      for (int k = 0; k < TOPK; ++k)
        if (topi[t * TOPK + k] == e) ++c;
    cnt_s[e] = c;
  }
  __syncthreads();
  if (e < Ee) {
    int b = 0;
    for (int j = 0; j < e; ++j) b += cnt_s[j];
    base[e] = b; counts[e] = cnt_s[e];
    int pos = 0;
    for (int t = 0; t < Tt; ++t)
#pragma unroll
      for (int k = 0; k < TOPK; ++k)
        if (topi[t * TOPK + k] == e) {
          tok_list[b + pos]       = t;
          slot_of[t * TOPK + k]   = b + pos;
          ++pos;
        }
  }
}

// ======================================================================
// Kernel 3: gate_up GEMM (bf16 WMMA, f32 accum) + clamp/GLU -> act (bf16)
// grid: (I/BNI=90, 8 m-tiles, E) ; block 256 = 8 waves
// wave w: N-subtile = w>>1, M-subtiles = 2*(w&1), 2*(w&1)+1  (2 WMMA / K-step)
// ======================================================================
__global__ __launch_bounds__(256) void gateup_kernel(
    const float* __restrict__ x,      // [T][H]
    const float* __restrict__ guw,    // [E][H][2I]
    const float* __restrict__ gub,    // [E][2I]
    const int* __restrict__ counts, const int* __restrict__ base,
    const int* __restrict__ tok_list, // [slot] -> token
    __bf16* __restrict__ act)         // [slot][I]
{
  const int e  = blockIdx.z;
  const int mt = blockIdx.y;
  const int nt = blockIdx.x;
  const int cnt = counts[e];
  const int m0  = mt * BM;
  if (m0 >= cnt) return;
  const int Mv  = min(BM, cnt - m0);
  const int bse = base[e];
  const int d0  = nt * BND;

  const int tid  = threadIdx.x;
  const int lane = tid & 31;
  const int wave = tid >> 5;
  const int ni  = wave >> 1;                 // 0..3 (16 d-cols each)
  const int ms0 = (wave & 1) * 2;            // m-subtiles ms0, ms0+1
  const int khalf = lane >> 4;
  const int lrow  = lane & 15;

  __shared__ int tokv[BM];
  __shared__ union {
    struct { __bf16 A[BM][LDA]; __bf16 Bt[BND][LDA]; } st;  // staging (bf16, B transposed)
    float C[BM][BND + 2];                                   // epilogue de-interleave
  } sm;

  if (tid < BM) {
    int p = m0 + tid;
    tokv[tid] = tok_list[bse + (p < cnt ? p : m0)];
  }
  __syncthreads();

  // ---- A staging: 64x32 f32 -> bf16 pairs along K. 1024 pairs, 4/thread.
  const float* axp[4]; unsigned* asp[4];
#pragma unroll
  for (int r = 0; r < 4; ++r) {
    int p = tid + r * 256;
    int m = p >> 4, kp = p & 15;                      // k = 2*kp
    axp[r] = x + (size_t)tokv[m] * Hh + 2 * kp;
    asp[r] = (unsigned*)&sm.st.A[m][2 * kp];
  }
  // ---- B staging: 32x64 f32 -> bf16 pairs along K (transposed). 1024 pairs, 4/thread.
  const float* bwp[4]; unsigned* bsp[4];
  const float* wbase = guw + (size_t)e * Hh * (2 * Ii) + d0;
  constexpr size_t WS = 2 * Ii;                       // weight row stride
#pragma unroll
  for (int r = 0; r < 4; ++r) {
    int p = tid + r * 256;
    int d = p & 63, kp = p >> 6;                      // k = 2*kp
    bwp[r] = wbase + (size_t)(2 * kp) * WS + d;
    bsp[r] = (unsigned*)&sm.st.Bt[d][2 * kp];
  }

  v8f acc0 = {0.f, 0.f, 0.f, 0.f, 0.f, 0.f, 0.f, 0.f};
  v8f acc1 = {0.f, 0.f, 0.f, 0.f, 0.f, 0.f, 0.f, 0.f};

  for (int k0 = 0; k0 < Hh; k0 += BK) {
#pragma unroll
    for (int r = 0; r < 4; ++r) {
      f32x2 v = *(const f32x2*)(axp[r] + k0);          // global_load_b64
      *asp[r] = pack_bf16(v[0], v[1]);                 // one cvt_pk + ds_store_b32
    }
#pragma unroll
    for (int r = 0; r < 4; ++r) {
      const float* pr = bwp[r] + (size_t)k0 * WS;
      *bsp[r] = pack_bf16(pr[0], pr[WS]);
    }
    __syncthreads();

    v16bf bfr = frag_b_lds(&sm.st.Bt[ni * 16 + lrow][0], khalf);
    v16bf a0  = frag_a_lds(&sm.st.A[(ms0 + 0) * 16 + lrow][0], khalf);
    v16bf a1  = frag_a_lds(&sm.st.A[(ms0 + 1) * 16 + lrow][0], khalf);
    acc0 = __builtin_amdgcn_wmma_f32_16x16x32_bf16(false, a0, false, bfr,
                                                   (short)0, acc0, false, false);
    acc1 = __builtin_amdgcn_wmma_f32_16x16x32_bf16(false, a1, false, bfr,
                                                   (short)0, acc1, false, false);
    __syncthreads();
  }

  // bias (per d-column, same for all rows of a lane)
  const int ccol = ni * 16 + lrow;
  float bias = gub[(size_t)e * (2 * Ii) + d0 + ccol];
#pragma unroll
  for (int r = 0; r < 8; ++r) { acc0[r] += bias; acc1[r] += bias; }

  // C layout: VGPR r -> row (r + 8*khalf), col = lane%16
#pragma unroll
  for (int r = 0; r < 8; ++r) {
    sm.C[(ms0 + 0) * 16 + khalf * 8 + r][ccol] = acc0[r];
    sm.C[(ms0 + 1) * 16 + khalf * 8 + r][ccol] = acc1[r];
  }
  __syncthreads();

  // de-interleave gate/up, activation, bf16 store (64x32 = 2048 vals, 8/thread)
#pragma unroll
  for (int r = 0; r < 8; ++r) {
    int idx = tid + r * 256;
    int m = idx >> 5, i = idx & 31;
    if (m < Mv) {
      float g = sm.C[m][2 * i];
      float u = sm.C[m][2 * i + 1];
      g = fminf(g, LIMIT);
      u = fminf(fmaxf(u, -LIMIT), LIMIT);
      float glu = g / (1.f + __expf(-g * ALPHA));
      float a = (u + 1.f) * glu;
      act[(size_t)(bse + m0 + m) * Ii + nt * BNI + i] = (__bf16)a;
    }
  }
}

// ======================================================================
// Kernel 4: down GEMM. A-tiles (bf16) via Tensor Data Mover -> LDS,
// weights f32->bf16 pair-staged, WMMA, eout[slot][H] = acc + down_b.
// grid: (H/BNH=45, 8 m-tiles, E) ; block 256 = 8 waves
// ======================================================================
__global__ __launch_bounds__(256) void down_kernel(
    const __bf16* __restrict__ act,   // [slot][I]
    const float* __restrict__ dw,     // [E][I][H]
    const float* __restrict__ db,     // [E][H]
    const int* __restrict__ counts, const int* __restrict__ base,
    float* __restrict__ eout)         // [slot][H]
{
  const int e  = blockIdx.z;
  const int mt = blockIdx.y;
  const int ht = blockIdx.x;
  const int cnt = counts[e];
  const int m0  = mt * BM;
  if (m0 >= cnt) return;
  const int Mv  = min(BM, cnt - m0);
  const int bse = base[e];
  const int h0  = ht * BNH;

  const int tid  = threadIdx.x;
  const int lane = tid & 31;
  const int wave = tid >> 5;
  const int ni  = wave >> 1;
  const int ms0 = (wave & 1) * 2;
  const int khalf = lane >> 4;
  const int lrow  = lane & 15;

  __shared__ alignas(16) __bf16 Atile[BM][LDA];   // TDM writes 64x32 + 1-DW row pad
  __shared__ alignas(16) __bf16 Btile[BNH][LDA];  // transposed down_w tile

  // ---- constant TDM descriptor group 1 (D#) ----
  // data_size=1 (2B), pad_enable=1, pad_interval=16DW (one 32-bf16 row), pad_amount=1DW
  i32x8 g1;
  g1[0] = (int)((1u << 16) | (1u << 20) | (3u << 22));
  g1[1] = (int)(((unsigned)Ii & 0xFFFFu) << 16);          // tensor_dim0[15:0]
  g1[2] = (int)(((unsigned)(Ee * Tt) & 0xFFFFu) << 16);   // dim0[31:16]=0 | tensor_dim1[15:0]
  g1[3] = (int)(32u << 16);                               // dim1[31:16]=0 | tile_dim0=32
  g1[4] = BM;                                             // tile_dim1=64 | tile_dim2=0
  g1[5] = (int)Ii;                                        // tensor_dim0_stride
  g1[6] = 0; g1[7] = 0;
  const i32x4 gz4 = {0, 0, 0, 0};
  const i32x8 gz8 = {0, 0, 0, 0, 0, 0, 0, 0};
  const unsigned ldsA = (unsigned)(uintptr_t)&Atile[0][0];
  const uint64_t arowbase =
      (uint64_t)(uintptr_t)act + (uint64_t)(bse + m0) * (uint64_t)Ii * 2ull;

  // ---- B staging: 32x64 f32 -> bf16 pairs along K (transposed). 1024 pairs, 4/thread.
  const float* bwp[4]; unsigned* bsp[4];
  const float* wbase = dw + (size_t)e * Ii * Hh + h0;
#pragma unroll
  for (int r = 0; r < 4; ++r) {
    int p = tid + r * 256;
    int d = p & 63, kp = p >> 6;
    bwp[r] = wbase + (size_t)(2 * kp) * Hh + d;
    bsp[r] = (unsigned*)&Btile[d][2 * kp];
  }

  v8f acc0 = {0.f, 0.f, 0.f, 0.f, 0.f, 0.f, 0.f, 0.f};
  v8f acc1 = {0.f, 0.f, 0.f, 0.f, 0.f, 0.f, 0.f, 0.f};

  for (int k0 = 0; k0 < Ii; k0 += BK) {
    // async: TDM pulls the 64x32 bf16 act tile into LDS (wave 0 issues once)
    if (tid < 32) {
      uint64_t ga = arowbase + (uint64_t)k0 * 2ull;
      u32x4 g0;
      g0[0] = 1u;                                    // count=1, user mode
      g0[1] = ldsA;                                  // LDS byte address
      g0[2] = (unsigned)ga;                          // global_addr[31:0]
      g0[3] = ((unsigned)(ga >> 32) & 0x01FFFFFFu) | (2u << 30);  // addr[56:32] | type=2
      __builtin_amdgcn_tensor_load_to_lds(g0, g1, gz4, gz4, gz8, 0);
    }
    // meanwhile all threads stage the f32 weight tile (packed cvt, transposed)
#pragma unroll
    for (int r = 0; r < 4; ++r) {
      const float* pr = bwp[r] + (size_t)k0 * Hh;
      *bsp[r] = pack_bf16(pr[0], pr[Hh]);
    }
    if (tid < 32) __builtin_amdgcn_s_wait_tensorcnt(0);
    __syncthreads();

    v16bf bfr = frag_b_lds(&Btile[ni * 16 + lrow][0], khalf);
    v16bf a0  = frag_a_lds(&Atile[(ms0 + 0) * 16 + lrow][0], khalf);
    v16bf a1  = frag_a_lds(&Atile[(ms0 + 1) * 16 + lrow][0], khalf);
    acc0 = __builtin_amdgcn_wmma_f32_16x16x32_bf16(false, a0, false, bfr,
                                                   (short)0, acc0, false, false);
    acc1 = __builtin_amdgcn_wmma_f32_16x16x32_bf16(false, a1, false, bfr,
                                                   (short)0, acc1, false, false);
    __syncthreads();
  }

  const int ccol = ni * 16 + lrow;
  const int h = h0 + ccol;
  float dbias = db[(size_t)e * Hh + h];
#pragma unroll
  for (int r = 0; r < 8; ++r) {
    int ma = (ms0 + 0) * 16 + khalf * 8 + r;
    int mb = (ms0 + 1) * 16 + khalf * 8 + r;
    if (ma < Mv) eout[(size_t)(bse + m0 + ma) * Hh + h] = acc0[r] + dbias;
    if (mb < Mv) eout[(size_t)(bse + m0 + mb) * Hh + h] = acc1[r] + dbias;
  }
}

// ======================================================================
// Kernel 5: deterministic gather: out[t][h] = sum_k topv[t][k]*eout[slot][h]
// ======================================================================
__global__ __launch_bounds__(256) void gather_kernel(
    const float* __restrict__ topv, const int* __restrict__ slot_of,
    const float* __restrict__ eout, float* __restrict__ out)
{
  int idx = blockIdx.x * 256 + threadIdx.x;
  if (idx >= Tt * Hh) return;
  int t = idx / Hh, h = idx % Hh;
  float s = 0.f;
#pragma unroll
  for (int k = 0; k < TOPK; ++k) {
    int slot = slot_of[t * TOPK + k];
    s += topv[t * TOPK + k] * eout[(size_t)slot * Hh + h];
  }
  out[idx] = s;
}

// ======================================================================
extern "C" void kernel_launch(void* const* d_in, const int* in_sizes, int n_in,
                              void* d_out, int out_size, void* d_ws, size_t ws_size,
                              hipStream_t stream) {
  const float* x   = (const float*)d_in[0];   // hidden_states [1][512][2880]
  const float* rw  = (const float*)d_in[1];   // router_w [H][E]
  const float* rb  = (const float*)d_in[2];   // router_b [E]
  const float* guw = (const float*)d_in[3];   // gate_up_w [E][H][2I]
  const float* gub = (const float*)d_in[4];   // gate_up_b [E][2I]
  const float* dw  = (const float*)d_in[5];   // down_w [E][I][H]
  const float* db  = (const float*)d_in[6];   // down_b [E][H]
  float* out = (float*)d_out;

  char* ws = (char*)d_ws;
  int*    topi     = (int*)(ws + OFF_TOPI);
  float*  topv     = (float*)(ws + OFF_TOPV);
  int*    counts   = (int*)(ws + OFF_CNT);
  int*    basep    = (int*)(ws + OFF_BASE);
  int*    tok_list = (int*)(ws + OFF_TOK);
  int*    slot_of  = (int*)(ws + OFF_SLOT);
  __bf16* act      = (__bf16*)(ws + OFF_ACT);
  float*  eout     = (float*)(ws + OFF_EOUT);

  router_kernel<<<Tt, 128, 0, stream>>>(x, rw, rb, topi, topv);
  build_lists_kernel<<<1, 16, 0, stream>>>(topi, counts, basep, tok_list, slot_of);
  gateup_kernel<<<dim3(Ii / BNI, MTILES, Ee), 256, 0, stream>>>(
      x, guw, gub, counts, basep, tok_list, act);
  down_kernel<<<dim3(Hh / BNH, MTILES, Ee), 256, 0, stream>>>(
      act, dw, db, counts, basep, eout);
  gather_kernel<<<(Tt * Hh + 255) / 256, 256, 0, stream>>>(topv, slot_of, eout, out);
}
